// RNNTLoss_11416023072991
// MI455X (gfx1250) — compile-verified, matrix operand order checked by reading, not run on hardware
//
#include <hip/hip_runtime.h>
#include <hip/hip_bf16.h>
#include <math.h>

// Problem constants from the reference: B=16, T=256, U=99, V=128
static constexpr int B_  = 16;
static constexpr int T_  = 256;
static constexpr int U_  = 99;
static constexpr int U1_ = 100;   // U + 1
static constexpr int V_  = 128;

typedef __attribute__((ext_vector_type(16))) __bf16 v16bf;
typedef __attribute__((ext_vector_type(8)))  float  v8f;

#define NEGINF (-INFINITY)

// -----------------------------------------------------------------------------
// Kernel 1: per-(b,t,u) logsumexp over V=128 via WMMA row-sum, then
//   blank_lp[row] = acts[row][0]        - (max + log(sum_exp))
//   emit_lp [row] = acts[row][label(u)] - (max + log(sum_exp))   (u < U)
//
// One wave handles 16 consecutive rows. Lane L = (j = L&15, h = L>>4):
// pair (L, L^16) owns row j; each half-lane holds 64 contiguous f32 of it.
// Sum of exp is computed with V_WMMA_F32_16X16X32_BF16 against an all-ones
// B matrix; hi/lo bf16 splitting keeps ~f32 accuracy (err ~2^-17 rel).
// -----------------------------------------------------------------------------
__global__ __launch_bounds__(256)
void rnnt_lp_kernel(const float* __restrict__ acts,
                    const int*   __restrict__ labels,
                    float*       __restrict__ blank_lp,
                    float*       __restrict__ emit_lp)
{
    const int tid  = blockIdx.x * 256 + threadIdx.x;
    const int wave = tid >> 5;
    const int lane = threadIdx.x & 31;
    const int j    = lane & 15;     // row within the wave's 16-row tile
    const int h    = lane >> 4;     // which 64-element half of the row

    const int row = wave * 16 + j;                 // global (b,t,u) row index
    const float* __restrict__ rp = acts + (size_t)row * V_;

    // ---- load this lane's 64 contiguous floats (16 x b128) -----------------
    const float4* __restrict__ p4 = (const float4*)(rp + h * 64);
    float4 x[16];
#pragma unroll
    for (int i = 0; i < 16; ++i) x[i] = p4[i];

    // ---- row max (lane-local, then pair with partner half-lane) -----------
    float m = NEGINF;
#pragma unroll
    for (int i = 0; i < 16; ++i)
        m = fmaxf(m, fmaxf(fmaxf(x[i].x, x[i].y), fmaxf(x[i].z, x[i].w)));
    m = fmaxf(m, __shfl_xor(m, 16, 32));           // row max shared by (L, L^16)

    // ---- sum of exp(x - m) via WMMA against all-ones B ---------------------
    v16bf ones;
#pragma unroll
    for (int k = 0; k < 16; ++k) ones[k] = (__bf16)1.0f;

    v8f c = {0.f, 0.f, 0.f, 0.f, 0.f, 0.f, 0.f, 0.f};

#pragma unroll
    for (int i = 0; i < 4; ++i) {                  // 4 K-chunks of 16 per lane
        v16bf ah, al;
#pragma unroll
        for (int k = 0; k < 4; ++k) {
            const float4 xv = x[i * 4 + k];
            const float e0 = __expf(xv.x - m);
            const float e1 = __expf(xv.y - m);
            const float e2 = __expf(xv.z - m);
            const float e3 = __expf(xv.w - m);
            const __bf16 h0 = (__bf16)e0, h1 = (__bf16)e1;
            const __bf16 h2 = (__bf16)e2, h3 = (__bf16)e3;
            ah[k * 4 + 0] = h0;  al[k * 4 + 0] = (__bf16)(e0 - (float)h0);
            ah[k * 4 + 1] = h1;  al[k * 4 + 1] = (__bf16)(e1 - (float)h1);
            ah[k * 4 + 2] = h2;  al[k * 4 + 2] = (__bf16)(e2 - (float)h2);
            ah[k * 4 + 3] = h3;  al[k * 4 + 3] = (__bf16)(e3 - (float)h3);
        }
        // D[m,n] += sum_k A[m,k] * 1 : accumulate per-row partial sums (f32)
        c = __builtin_amdgcn_wmma_f32_16x16x32_bf16(false, ah, false, ones,
                                                    (short)0, c, false, false);
        c = __builtin_amdgcn_wmma_f32_16x16x32_bf16(false, al, false, ones,
                                                    (short)0, c, false, false);
    }

    // ---- broadcast row sums: c[q]@lane0 = row q, c[q]@lane16 = row q+8 ----
    float s = 0.f;
#pragma unroll
    for (int q = 0; q < 8; ++q) {
        const float a0 = __shfl(c[q], 0, 32);
        const float a1 = __shfl(c[q], 16, 32);
        if (j == q)     s = a0;
        if (j == q + 8) s = a1;
    }

    const float lse = m + __logf(s);               // logsumexp of this row

    if (h == 0) {
        blank_lp[row] = rp[0] - lse;               // BLANK == 0
        const int b = row / (T_ * U1_);
        const int u = row % U1_;
        if (u < U_) {
            const int lbl = labels[b * U_ + u];
            emit_lp[row] = rp[lbl] - lse;
        }
    }
}

// -----------------------------------------------------------------------------
// Kernel 2: forward DP over the (T, U+1) lattice, one wave32 per example.
// Anti-diagonal wavefront: on diagonal d, cell (t,u) with t = d - u needs
//   up   = alpha[t-1][u]   + blank_lp[t-1][u]   (prev diagonal, same slot)
//   left = alpha[t][u-1]   + emit_lp[t][u-1]    (prev diagonal, slot u-1)
// Lane L owns u = 4L..4L+3; slot u-1 crosses lanes only at jj==0 (shfl_up).
// alpha[0][0] = 0; boundaries fall out of logaddexp with -inf operands.
// -----------------------------------------------------------------------------
__device__ __forceinline__ float log_add_exp(float a, float b)
{
    const float mx = fmaxf(a, b);
    if (mx == NEGINF) return NEGINF;
    return mx + log1pf(__expf(fminf(a, b) - mx));
}

__global__ __launch_bounds__(32)
void rnnt_dp_kernel(const float* __restrict__ blank_lp,
                    const float* __restrict__ emit_lp,
                    const int*   __restrict__ act_lens,
                    const int*   __restrict__ label_lens,
                    float*       __restrict__ costs)
{
    const int b    = blockIdx.x;
    const int lane = threadIdx.x;

    const float* __restrict__ bp = blank_lp + (size_t)b * T_ * U1_;
    const float* __restrict__ ep = emit_lp  + (size_t)b * T_ * U1_;

    const int t_len    = act_lens[b];
    const int u_len    = label_lens[b];
    const int target_d = (t_len - 1) + u_len;

    float prev[4] = {NEGINF, NEGINF, NEGINF, NEGINF};

    for (int d = 0; d <= (T_ - 1) + U_; ++d) {
        float nb = __shfl_up(prev[3], 1, 32);      // prev-diag value at u-1
        if (lane == 0) nb = NEGINF;

        float cur[4];
#pragma unroll
        for (int jj = 0; jj < 4; ++jj) {
            const int u = lane * 4 + jj;
            const int t = d - u;
            const bool valid = (u < U1_) && (t >= 0) && (t < T_);

            float up = NEGINF, left = NEGINF;
            if (valid && t >= 1) up = prev[jj] + bp[(t - 1) * U1_ + u];
            const float pl = jj ? prev[jj - 1] : nb;
            if (valid && u >= 1) left = pl + ep[t * U1_ + (u - 1)];

            float a;
            if (t == 0 && u == 0) a = 0.0f;        // alpha[0][0]
            else                  a = log_add_exp(up, left);
            cur[jj] = valid ? a : NEGINF;

            // Prefetch next diagonal's operands (global_prefetch_b8);
            // addresses are data-independent, hiding L2/L0 latency.
            const int t2 = t + 1;                  // = (d+1) - u
            if (u < U1_ && t2 >= 1 && t2 < T_)
                __builtin_prefetch(&bp[(t2 - 1) * U1_ + u], 0, 3);
            if (u >= 1 && u < U1_ && t2 >= 0 && t2 < T_)
                __builtin_prefetch(&ep[t2 * U1_ + (u - 1)], 0, 3);
        }

        if (d == target_d) {
#pragma unroll
            for (int jj = 0; jj < 4; ++jj)
                if (lane * 4 + jj == u_len)
                    costs[b] = -(cur[jj] + bp[(t_len - 1) * U1_ + u_len]);
        }

#pragma unroll
        for (int jj = 0; jj < 4; ++jj) prev[jj] = cur[jj];
    }
}

// -----------------------------------------------------------------------------
// Kernel 3: deterministic order-fixed reduction of the 16 per-example costs.
// -----------------------------------------------------------------------------
__global__ void rnnt_finalize_kernel(const float* __restrict__ costs,
                                     float* __restrict__ out)
{
    if (threadIdx.x == 0 && blockIdx.x == 0) {
        float s = 0.f;
        for (int i = 0; i < B_; ++i) s += costs[i];
        out[0] = s / (float)B_;
    }
}

// -----------------------------------------------------------------------------
extern "C" void kernel_launch(void* const* d_in, const int* in_sizes, int n_in,
                              void* d_out, int out_size, void* d_ws, size_t ws_size,
                              hipStream_t stream)
{
    (void)in_sizes; (void)n_in; (void)out_size; (void)ws_size;

    const float* acts       = (const float*)d_in[0];
    const int*   labels     = (const int*)  d_in[1];
    const int*   act_lens   = (const int*)  d_in[2];
    const int*   label_lens = (const int*)  d_in[3];
    float*       out        = (float*)d_out;

    // workspace layout (floats): blank_lp | emit_lp | per-example costs
    float* ws       = (float*)d_ws;
    float* blank_lp = ws;
    float* emit_lp  = ws + (size_t)B_ * T_ * U1_;
    float* costs    = emit_lp + (size_t)B_ * T_ * U1_;

    // 409,600 rows, 16 rows per wave, 8 waves per block -> 3200 blocks exact.
    const int rows   = B_ * T_ * U1_;
    const int blocks = rows / (16 * 8);

    rnnt_lp_kernel<<<blocks, 256, 0, stream>>>(acts, labels, blank_lp, emit_lp);
    rnnt_dp_kernel<<<B_, 32, 0, stream>>>(blank_lp, emit_lp, act_lens,
                                          label_lens, costs);
    rnnt_finalize_kernel<<<1, 32, 0, stream>>>(costs, out);
}
